// ImprovedYoloV8s_67929202753867
// MI455X (gfx1250) — matrix operand order repaired and checked by reading, not compiled, not run on hardware
//
#include <hip/hip_runtime.h>

// ---------------------------------------------------------------------------
// CDNA5 / gfx1250 implementation of ImprovedYoloV8s forward.
// Convs = implicit GEMM on v_wmma_f32_16x16x32_bf16 (wave32), f32 accumulate.
// Templated on kernel size (1/3) and N-tiles per wave (1/2/4) so one im2col
// A-gather feeds up to 4 WMMAs.  Epilogue transposes the D tile through LDS
// so stores are coalesced global_store_b128.  EMA recurrence = wave-parallel
// affine scan via __shfl_up (wave32).
// ---------------------------------------------------------------------------

typedef __bf16 bf16;
typedef __attribute__((ext_vector_type(16))) __bf16 v16bf;
typedef __attribute__((ext_vector_type(8)))  float  v8f;
typedef __attribute__((ext_vector_type(4)))  float  v4f;

#define ACT_SILU 0
#define ACT_RELU 1
#define ACT_NONE 2

__device__ __forceinline__ float act_apply(float v, int act) {
  if (act == ACT_SILU) return v / (1.f + __expf(-v));
  if (act == ACT_RELU) return fmaxf(v, 0.f);
  return v;
}

// ---- weight packing: OIHW fp32 * bn_scale -> [Cout][Kpad] bf16 -------------
__global__ __launch_bounds__(256) void pack_w_kernel(
    const float* __restrict__ w, const float* __restrict__ s,
    bf16* __restrict__ wp, int K, int Kpad, int total) {
  int idx = blockIdx.x * blockDim.x + threadIdx.x;
  if (idx >= total) return;
  int co = idx / Kpad, k = idx - co * Kpad;
  float v = (k < K) ? w[(size_t)co * K + k] * s[co] : 0.f;
  wp[idx] = (bf16)v;
}

// ---- implicit-GEMM conv ----------------------------------------------------
// Per wave: one 16-row (spatial) A tile feeds NT 16x16 WMMA tiles (NT*16 couts).
// A (16x32 bf16) per ISA layout: lanes 0-15 own K {0..7,16..23}, lanes 16-31
// own K {8..15,24..31}.  B lanes hold the same K set from packed weights.
// Note: M = B*OH*OW and Cout are always multiples of 16 in this network, and
// OH*OW is a multiple of 16, so a 16-row M tile is contiguous within one batch.
template <int KSZ, int NT>
__global__ __launch_bounds__(128) void conv_wmma_t(
    const float* __restrict__ in, const bf16* __restrict__ wp,
    const float* __restrict__ bias, float* __restrict__ out,
    int B, int Cin, int Cintot, int ci0, int H, int W,
    int Cout, int stride, int OH, int OW, int Ctot, int co0, int act) {
  const int lane = threadIdx.x & 31;
  const int wv   = threadIdx.x >> 5;
  const int M    = B * OH * OW;
  const int mtile = blockIdx.x;
  const int ntb   = (blockIdx.y * 4 + wv) * NT;  // first n-tile of this wave
  if (ntb * 16 >= Cout) return;                  // wave-uniform exit

  constexpr int KK  = KSZ * KSZ;
  constexpr int PAD = KSZ >> 1;
  const int K    = Cin * KK;
  const int Kpad = (K + 31) & ~31;

  // A-operand row (M) for this lane
  const int mrow = mtile * 16 + (lane & 15);
  const bool mvalid = (mrow < M);
  int b = 0, oh = 0, ow = 0;
  if (mvalid) {
    b = mrow / (OH * OW);
    int r = mrow - b * (OH * OW);
    oh = r / OW; ow = r - oh * OW;
  }
  const int kbaseA = (lane < 16) ? 0 : 8;
  const float* inb = in + ((size_t)b * Cintot + ci0) * H * W;
  const int ihb = oh * stride - PAD, iwb = ow * stride - PAD;

  // B-operand row pointers (one 16-cout tile each)
  const bf16* brow[NT];
#pragma unroll
  for (int t = 0; t < NT; ++t) {
    int ncol = (ntb + t) * 16 + (lane & 15);
    if (ncol >= Cout) ncol = Cout - 1;
    brow[t] = wp + (size_t)ncol * Kpad + ((lane >> 4) << 3);
  }

  v8f acc[NT] = {};
  for (int k0 = 0; k0 < Kpad; k0 += 32) {
    v16bf a;
#pragma unroll
    for (int i = 0; i < 16; ++i) {
      int k = k0 + kbaseA + i + ((i >= 8) ? 8 : 0);
      float v = 0.f;
      if (mvalid && k < K) {
        if (KSZ == 1) {
          v = inb[((size_t)k * H + oh * stride) * W + ow * stride];
        } else {
          int cin = k / KK;           // constant divisor -> mul/shift
          int r   = k - cin * KK;
          int ky  = r / KSZ, kx = r - ky * KSZ;
          int ih  = ihb + ky, iw = iwb + kx;
          if ((unsigned)ih < (unsigned)H && (unsigned)iw < (unsigned)W)
            v = inb[((size_t)cin * H + ih) * W + iw];
        }
      }
      a[i] = (bf16)v;
    }
    if (k0 + 32 < Kpad)  // stream next weight chunk toward L0/L2
      __builtin_prefetch((const void*)(brow[0] + k0 + 32), 0, 1);
#pragma unroll
    for (int t = 0; t < NT; ++t) {
      if ((ntb + t) * 16 < Cout) {  // wave-uniform
        v16bf bm;
#pragma unroll
        for (int i = 0; i < 16; ++i) bm[i] = brow[t][k0 + i + ((i >= 8) ? 8 : 0)];
        acc[t] = __builtin_amdgcn_wmma_f32_16x16x32_bf16(
            false, a, false, bm, (short)0, acc[t], false, false);
      }
    }
  }

  // ---- epilogue: bias+act, transpose 16x16 tile through LDS, store b128 ----
  // D layout in: column n = lane&15; VGPR r -> M = r (+8 for lanes 16..31).
  // LDS row = cout (stride 17 words for bank spread), col = spatial m.
  __shared__ float ldst[4][16 * 17];
  float* lbuf = ldst[wv];
  const int OHOW = OH * OW;
  const int mg0  = mtile * 16;
  const int bb   = mg0 / OHOW;
  const int r0   = mg0 - bb * OHOW;  // 16-aligned, contiguous within batch bb
  const int nlo  = lane & 15;
  const int mh   = (lane >> 4) * 8;
#pragma unroll
  for (int t = 0; t < NT; ++t) {
    if ((ntb + t) * 16 >= Cout) continue;
    float bval = bias[(ntb + t) * 16 + nlo];
#pragma unroll
    for (int r = 0; r < 8; ++r) {
      int m = (lane < 16) ? r : (r + 8);
      lbuf[nlo * 17 + m] = act_apply(acc[t][r] + bval, act);
    }
    // DS is in-order within a wave: safe to read the transpose back now.
    v4f s0, s1;
#pragma unroll
    for (int j = 0; j < 4; ++j) s0[j] = lbuf[nlo * 17 + mh + j];
#pragma unroll
    for (int j = 0; j < 4; ++j) s1[j] = lbuf[nlo * 17 + mh + 4 + j];
    float* op = out + ((size_t)bb * Ctot + co0 + (ntb + t) * 16 + nlo) * OHOW + r0 + mh;
    *(v4f*)op       = s0;
    *((v4f*)op + 1) = s1;
  }
}

// ---- EMA attention: depthwise conv1d(k=3) + IIR scan + residual ------------
// One wave per (b,c) row.  ema_i = a_i*ema_{i-1} + b_i with a_0=0;
// lanes compose their L/32 segment into an affine map, __shfl_up scan
// composes across lanes, second pass emits conv + ema + x.
__global__ __launch_bounds__(256) void ema_kernel(
    const float* __restrict__ x, const float* __restrict__ w,
    const float* __restrict__ b, float* __restrict__ out,
    int B, int C, int L, int Ctot, int co0) {
  int wave = (blockIdx.x * blockDim.x + threadIdx.x) >> 5;
  int lane = threadIdx.x & 31;
  if (wave >= B * C) return;
  int bb = wave / C, c = wave - bb * C;
  const float* xr = x + (size_t)wave * L;
  float* yr = out + ((size_t)bb * Ctot + co0 + c) * L;
  float w0 = w[3 * c], w1 = w[3 * c + 1], w2 = w[3 * c + 2], bc = b[c];

  const int seg = L >> 5;  // L is a power of two >= 256
  const int s0  = lane * seg;

  // pass 1: compose segment affine map  e_out = A*e_in + Bv
  float A = 1.f, Bv = 0.f;
  for (int i = 0; i < seg; ++i) {
    int gi = s0 + i;
    float xi = xr[gi];
    float ai = (gi == 0) ? 0.f : 0.9f;
    float bi = (gi == 0) ? xi : 0.1f * xi;
    Bv = ai * Bv + bi;
    A  = ai * A;
  }
  // inclusive wave scan of affine composition (later-after-earlier)
  for (int off = 1; off < 32; off <<= 1) {
    float uA = __shfl_up(A, off, 32);
    float uB = __shfl_up(Bv, off, 32);
    if (lane >= off) { Bv = A * uB + Bv; A = A * uA; }
  }
  // exclusive prefix: ema value entering this segment (a_0=0 kills e_init)
  float eB = __shfl_up(Bv, 1, 32);
  float e  = (lane == 0) ? 0.f : eB;

  // pass 2: emit outputs
  for (int i = 0; i < seg; ++i) {
    int gi = s0 + i;
    float xi = xr[gi];
    float xm = (gi > 0) ? xr[gi - 1] : 0.f;
    float xp = (gi + 1 < L) ? xr[gi + 1] : 0.f;
    e = (gi == 0) ? xi : (0.1f * xi + 0.9f * e);
    yr[gi] = w0 * xm + w1 * xi + w2 * xp + bc + e + xi;
  }
}

// ---- 5x5 stride-1 pad-2 maxpool on a channel slice of a concat buffer ------
__global__ __launch_bounds__(256) void maxpool5_kernel(
    const float* __restrict__ in, float* __restrict__ out,
    int B, int Ctot, int ci0, int co0, int C, int H, int W) {
  int idx = blockIdx.x * blockDim.x + threadIdx.x;
  int total = B * C * H * W;
  if (idx >= total) return;
  int w = idx % W, h = (idx / W) % H, c = (idx / (W * H)) % C, b = idx / (W * H * C);
  const float* p = in + ((size_t)b * Ctot + ci0 + c) * H * W;
  float m = -3.402823466e38f;
  for (int dy = -2; dy <= 2; ++dy) {
    int y = h + dy; if ((unsigned)y >= (unsigned)H) continue;
    for (int dx = -2; dx <= 2; ++dx) {
      int xw = w + dx; if ((unsigned)xw >= (unsigned)W) continue;
      m = fmaxf(m, p[y * W + xw]);
    }
  }
  out[((size_t)b * Ctot + co0 + c) * H * W + h * W + w] = m;
}

// ---- copy a tensor into a channel slice of a concat buffer -----------------
__global__ __launch_bounds__(256) void copy_into_kernel(
    const float* __restrict__ in, float* __restrict__ out,
    int B, int C, int HW, int Ctot, int co0) {
  int idx = blockIdx.x * blockDim.x + threadIdx.x;
  if (idx >= B * C * HW) return;
  int hw = idx % HW, c = (idx / HW) % C, b = idx / (HW * C);
  out[((size_t)b * Ctot + co0 + c) * HW + hw] = in[idx];
}

// ---- nearest 2x upsample into a channel slice of a concat buffer -----------
__global__ __launch_bounds__(256) void up2_into_kernel(
    const float* __restrict__ in, float* __restrict__ out,
    int B, int C, int H, int W, int Ctot, int co0) {
  int OH = 2 * H, OW = 2 * W;
  int idx = blockIdx.x * blockDim.x + threadIdx.x;
  if (idx >= B * C * OH * OW) return;
  int ow = idx % OW, oh = (idx / OW) % OH, c = (idx / (OW * OH)) % C, b = idx / (OW * OH * C);
  out[(((size_t)b * Ctot + co0 + c) * OH + oh) * OW + ow] =
      in[(((size_t)b * C + c) * H + (oh >> 1)) * W + (ow >> 1)];
}

// ---- detect head helpers ---------------------------------------------------
__global__ __launch_bounds__(256) void rowmean_kernel(
    const float* __restrict__ f, float* __restrict__ g, int BC, int HW) {
  int wave = (blockIdx.x * blockDim.x + threadIdx.x) >> 5;
  int lane = threadIdx.x & 31;
  if (wave >= BC) return;
  const float* p = f + (size_t)wave * HW;
  float s = 0.f;
  for (int i = lane; i < HW; i += 32) s += p[i];
  for (int off = 16; off; off >>= 1) s += __shfl_down(s, off, 32);
  if (lane == 0) g[wave] = s / (float)HW;
}

__global__ __launch_bounds__(64) void mlp_kernel(
    const float* __restrict__ in, const float* __restrict__ w,
    const float* __restrict__ bias, float* __restrict__ out,
    int B, int IN, int OUT, int mode /*0=relu,1=sigmoid*/) {
  int idx = blockIdx.x * blockDim.x + threadIdx.x;
  if (idx >= B * OUT) return;
  int b = idx / OUT, o = idx - b * OUT;
  const float* wr = w + (size_t)o * IN;
  const float* ir = in + (size_t)b * IN;
  float s = bias[o];
  for (int i = 0; i < IN; ++i) s += ir[i] * wr[i];
  out[idx] = (mode == 0) ? fmaxf(s, 0.f) : 1.f / (1.f + __expf(-s));
}

// y[b,h,w,o] = sum_c f[b,c,h,w]*attn[b,c]*cw[o,c] + cb[o]   ([B,1,H,W,no])
__global__ __launch_bounds__(256) void det_out_kernel(
    const float* __restrict__ f, const float* __restrict__ attn,
    const float* __restrict__ cw, const float* __restrict__ cb,
    float* __restrict__ out, int B, int C, int H, int W, int NO) {
  int idx = blockIdx.x * blockDim.x + threadIdx.x;
  int total = B * H * W * NO;
  if (idx >= total) return;
  int o = idx % NO, w = (idx / NO) % W, h = (idx / (NO * W)) % H, b = idx / (NO * W * H);
  float s = cb[o];
  const float* ar = attn + (size_t)b * C;
  for (int c = 0; c < C; ++c)
    s += f[(((size_t)b * C + c) * H + h) * W + w] * ar[c] * cw[(size_t)o * C + c];
  out[idx] = s;
}

// ===========================================================================
// Host orchestration
// ===========================================================================
namespace {

struct Arena {
  char* p; size_t off; size_t cap;
  void* alloc(size_t bytes) {
    size_t o = (off + 255) & ~(size_t)255;
    off = o + bytes;
    return p + o;
  }
};
static float* AF(Arena& a, size_t elems) { return (float*)a.alloc(elems * sizeof(float)); }
static inline unsigned cdiv(unsigned a, unsigned b) { return (a + b - 1) / b; }

struct PackedConv {
  const bf16* wp; const float* bias; int Cin, Cout, k, Kpad;
};

// conv param dict flattens (sorted keys) as: b, s, w  at base+0,+1,+2
static PackedConv pack_conv(void* const* din, int base, int Cin, int Cout, int k,
                            Arena& ar, hipStream_t st) {
  int K = Cin * k * k, Kpad = (K + 31) & ~31;
  bf16* wp = (bf16*)ar.alloc((size_t)Cout * Kpad * sizeof(bf16));
  int total = Cout * Kpad;
  pack_w_kernel<<<cdiv(total, 256), 256, 0, st>>>(
      (const float*)din[base + 2], (const float*)din[base + 1], wp, K, Kpad, total);
  PackedConv pc; pc.wp = wp; pc.bias = (const float*)din[base + 0];
  pc.Cin = Cin; pc.Cout = Cout; pc.k = k; pc.Kpad = Kpad;
  return pc;
}

static void conv_run(const PackedConv& pc, const float* in, int Cintot, int ci0,
                     int B, int H, int W, int stride, float* out, int Ctot,
                     int co0, int act, hipStream_t st) {
  int OH = H / stride, OW = W / stride;  // 'same' pad, even H/W
  // NT n-tiles per wave (pick so every wave in the last block stays busy)
  int NT = (pc.Cout % 256 == 0) ? 4 : ((pc.Cout % 128 == 0) ? 2 : 1);
  dim3 grid(cdiv((unsigned)(B * OH * OW), 16), cdiv((unsigned)pc.Cout, 64u * NT));
#define CONV_ARGS in, pc.wp, pc.bias, out, B, pc.Cin, Cintot, ci0, H, W, \
                  pc.Cout, stride, OH, OW, Ctot, co0, act
  if (pc.k == 1) {
    if (NT == 4)      conv_wmma_t<1, 4><<<grid, 128, 0, st>>>(CONV_ARGS);
    else if (NT == 2) conv_wmma_t<1, 2><<<grid, 128, 0, st>>>(CONV_ARGS);
    else              conv_wmma_t<1, 1><<<grid, 128, 0, st>>>(CONV_ARGS);
  } else {
    if (NT == 4)      conv_wmma_t<3, 4><<<grid, 128, 0, st>>>(CONV_ARGS);
    else if (NT == 2) conv_wmma_t<3, 2><<<grid, 128, 0, st>>>(CONV_ARGS);
    else              conv_wmma_t<3, 1><<<grid, 128, 0, st>>>(CONV_ARGS);
  }
#undef CONV_ARGS
}

// C2f_Faster_EMA: flatten order cv1(b,s,w) cv2(b,s,w) ema_b ema_w m0(..) m1(..)
static void c2f_run(void* const* din, int base, const float* x, int Cin, int c2,
                    int B, int H, int W, float* out, float* cat, float* t1,
                    float* t2, Arena& ar, hipStream_t st) {
  const int c = c2 / 2, HW = H * W;
  PackedConv cv1 = pack_conv(din, base + 0, Cin, 2 * c, 1, ar, st);
  PackedConv cv2 = pack_conv(din, base + 3, 4 * c, c2, 1, ar, st);
  const float* ema_b = (const float*)din[base + 6];
  const float* ema_w = (const float*)din[base + 7];
  PackedConv m0a = pack_conv(din, base + 8,  c, c, 3, ar, st);
  PackedConv m0b = pack_conv(din, base + 11, c, c, 3, ar, st);
  PackedConv m1a = pack_conv(din, base + 14, c, c, 3, ar, st);
  PackedConv m1b = pack_conv(din, base + 17, c, c, 3, ar, st);

  const unsigned ema_thr = (unsigned)(B * c) * 32u;
  conv_run(cv1, x, Cin, 0, B, H, W, 1, cat, 4 * c, 0, ACT_SILU, st);
  // bottleneck 0 (input = cat channels [c, 2c))
  conv_run(m0a, cat, 4 * c, c, B, H, W, 1, t1, c, 0, ACT_SILU, st);
  conv_run(m0b, t1, c, 0, B, H, W, 1, t2, c, 0, ACT_SILU, st);
  ema_kernel<<<cdiv(ema_thr, 256), 256, 0, st>>>(t2, ema_w, ema_b, cat, B, c, HW, 4 * c, 2 * c);
  // bottleneck 1 (input = cat channels [2c, 3c))
  conv_run(m1a, cat, 4 * c, 2 * c, B, H, W, 1, t1, c, 0, ACT_SILU, st);
  conv_run(m1b, t1, c, 0, B, H, W, 1, t2, c, 0, ACT_SILU, st);
  ema_kernel<<<cdiv(ema_thr, 256), 256, 0, st>>>(t2, ema_w, ema_b, cat, B, c, HW, 4 * c, 3 * c);
  conv_run(cv2, cat, 4 * c, 0, B, H, W, 1, out, c2, 0, ACT_SILU, st);
}

// detect dict flattens as: cv_b, cv_w, sa_b1, sa_b2, sa_w1, sa_w2
static void det_run(void* const* din, int base, const float* f, int C, int H,
                    int W, float* out, Arena& ar, int B, hipStream_t st) {
  const int mid = C / 4, no = 9, HW = H * W;
  float* g    = AF(ar, (size_t)B * C);
  float* hid  = AF(ar, (size_t)B * mid);
  float* attn = AF(ar, (size_t)B * C);
  rowmean_kernel<<<cdiv((unsigned)(B * C) * 32u, 256), 256, 0, st>>>(f, g, B * C, HW);
  mlp_kernel<<<cdiv(B * mid, 64), 64, 0, st>>>(
      g, (const float*)din[base + 4], (const float*)din[base + 2], hid, B, C, mid, 0);
  mlp_kernel<<<cdiv(B * C, 64), 64, 0, st>>>(
      hid, (const float*)din[base + 5], (const float*)din[base + 3], attn, B, mid, C, 1);
  int total = B * HW * no;
  det_out_kernel<<<cdiv(total, 256), 256, 0, st>>>(
      f, attn, (const float*)din[base + 1], (const float*)din[base + 0],
      out, B, C, H, W, no);
}

}  // namespace

extern "C" void kernel_launch(void* const* d_in, const int* in_sizes, int n_in,
                              void* d_out, int out_size, void* d_ws, size_t ws_size,
                              hipStream_t stream) {
  (void)in_sizes; (void)n_in; (void)out_size;
  const int B = 4;
  Arena ar{(char*)d_ws, 0, ws_size};

  // --- flat input indices (jax pytree sorted-key flatten of {params, x}) ---
  enum {
    I_B0 = 0,
    I_B1_C2F = 3,   I_B1_CONV = 23,
    I_B2_C2F = 26,  I_B2_CONV = 46,
    I_B3_C2F = 49,  I_B3_CONV = 69,
    I_B4_C2F = 72,  I_B4_CONV = 92, I_SPPF_CV1 = 95, I_SPPF_CV2 = 98,
    I_DET0 = 101,   I_DET1 = 107,   I_DET2 = 113,
    I_N0 = 119,
    I_N1_C2F = 122, I_N1_CONV = 142,
    I_N2_C2F = 145,
    I_N3 = 165,
    I_N4_C2F = 168, I_N4_CONV = 188,
    I_N5_C2F = 191,
    I_X = 211
  };
  const float* x = (const float*)d_in[I_X];  // [4,3,512,512]

  // --- activation arena (working set ~180MB -> L2-resident on MI455X) ------
  float* X1   = AF(ar, 4ull * 32 * 256 * 256);
  float* T    = AF(ar, 4ull * 64 * 128 * 128);   // pre-C2f conv outs (reused)
  float* CAT  = AF(ar, 4ull * 128 * 128 * 128);  // C2f concat (max 4c*HW)
  float* T1   = AF(ar, 4ull * 32 * 128 * 128);
  float* T2   = AF(ar, 4ull * 32 * 128 * 128);
  float* X2   = AF(ar, 4ull * 64 * 128 * 128);
  float* X3   = AF(ar, 4ull * 128 * 64 * 64);
  float* X4   = AF(ar, 4ull * 256 * 32 * 32);
  float* U    = AF(ar, 4ull * 512 * 16 * 16);
  float* SCAT = AF(ar, 4ull * 1024 * 16 * 16);
  float* X5   = AF(ar, 4ull * 512 * 16 * 16);
  float* N0O  = AF(ar, 4ull * 256 * 16 * 16);
  float* P4C  = AF(ar, 4ull * 512 * 32 * 32);
  float* N1O  = AF(ar, 4ull * 256 * 32 * 32);
  float* N1C  = AF(ar, 4ull * 128 * 32 * 32);
  float* P3C  = AF(ar, 4ull * 256 * 64 * 64);
  float* P3   = AF(ar, 4ull * 128 * 64 * 64);
  float* P4C2 = AF(ar, 4ull * 384 * 32 * 32);
  float* N4O  = AF(ar, 4ull * 256 * 32 * 32);
  float* P5C  = AF(ar, 4ull * 768 * 16 * 16);
  float* P5   = AF(ar, 4ull * 512 * 16 * 16);

  // ---------------- backbone ----------------
  PackedConv b0 = pack_conv(d_in, I_B0, 3, 32, 3, ar, stream);
  conv_run(b0, x, 3, 0, B, 512, 512, 2, X1, 32, 0, ACT_SILU, stream);

  PackedConv b1c = pack_conv(d_in, I_B1_CONV, 32, 64, 3, ar, stream);
  conv_run(b1c, X1, 32, 0, B, 256, 256, 2, T, 64, 0, ACT_SILU, stream);
  c2f_run(d_in, I_B1_C2F, T, 64, 64, B, 128, 128, X2, CAT, T1, T2, ar, stream);

  PackedConv b2c = pack_conv(d_in, I_B2_CONV, 64, 128, 3, ar, stream);
  conv_run(b2c, X2, 64, 0, B, 128, 128, 2, T, 128, 0, ACT_SILU, stream);
  c2f_run(d_in, I_B2_C2F, T, 128, 128, B, 64, 64, X3, CAT, T1, T2, ar, stream);

  PackedConv b3c = pack_conv(d_in, I_B3_CONV, 128, 256, 3, ar, stream);
  conv_run(b3c, X3, 128, 0, B, 64, 64, 2, T, 256, 0, ACT_SILU, stream);
  c2f_run(d_in, I_B3_C2F, T, 256, 256, B, 32, 32, X4, CAT, T1, T2, ar, stream);

  PackedConv b4c = pack_conv(d_in, I_B4_CONV, 256, 512, 3, ar, stream);
  conv_run(b4c, X4, 256, 0, B, 32, 32, 2, T, 512, 0, ACT_SILU, stream);
  c2f_run(d_in, I_B4_C2F, T, 512, 512, B, 16, 16, U, CAT, T1, T2, ar, stream);

  // SPPF (SimSPPF: ReLU)
  PackedConv scv1 = pack_conv(d_in, I_SPPF_CV1, 512, 256, 1, ar, stream);
  conv_run(scv1, U, 512, 0, B, 16, 16, 1, SCAT, 1024, 0, ACT_RELU, stream);
  {
    int n = B * 256 * 16 * 16;
    maxpool5_kernel<<<cdiv(n, 256), 256, 0, stream>>>(SCAT, SCAT, B, 1024, 0,   256, 256, 16, 16);
    maxpool5_kernel<<<cdiv(n, 256), 256, 0, stream>>>(SCAT, SCAT, B, 1024, 256, 512, 256, 16, 16);
    maxpool5_kernel<<<cdiv(n, 256), 256, 0, stream>>>(SCAT, SCAT, B, 1024, 512, 768, 256, 16, 16);
  }
  PackedConv scv2 = pack_conv(d_in, I_SPPF_CV2, 1024, 512, 1, ar, stream);
  conv_run(scv2, SCAT, 1024, 0, B, 16, 16, 1, X5, 512, 0, ACT_RELU, stream);

  // ---------------- neck (FPN/PAN) ----------------
  PackedConv n0 = pack_conv(d_in, I_N0, 512, 256, 1, ar, stream);
  conv_run(n0, X5, 512, 0, B, 16, 16, 1, N0O, 256, 0, ACT_SILU, stream);
  up2_into_kernel<<<cdiv(B * 256 * 32 * 32, 256), 256, 0, stream>>>(N0O, P4C, B, 256, 16, 16, 512, 0);
  copy_into_kernel<<<cdiv(B * 256 * 32 * 32, 256), 256, 0, stream>>>(X4, P4C, B, 256, 32 * 32, 512, 256);
  c2f_run(d_in, I_N1_C2F, P4C, 512, 256, B, 32, 32, N1O, CAT, T1, T2, ar, stream);

  PackedConv n1c = pack_conv(d_in, I_N1_CONV, 256, 128, 1, ar, stream);
  conv_run(n1c, N1O, 256, 0, B, 32, 32, 1, N1C, 128, 0, ACT_SILU, stream);
  up2_into_kernel<<<cdiv(B * 128 * 64 * 64, 256), 256, 0, stream>>>(N1C, P3C, B, 128, 32, 32, 256, 0);
  copy_into_kernel<<<cdiv(B * 128 * 64 * 64, 256), 256, 0, stream>>>(X3, P3C, B, 128, 64 * 64, 256, 128);
  c2f_run(d_in, I_N2_C2F, P3C, 256, 128, B, 64, 64, P3, CAT, T1, T2, ar, stream);

  PackedConv n3 = pack_conv(d_in, I_N3, 128, 128, 3, ar, stream);
  conv_run(n3, P3, 128, 0, B, 64, 64, 2, P4C2, 384, 0, ACT_SILU, stream);
  copy_into_kernel<<<cdiv(B * 256 * 32 * 32, 256), 256, 0, stream>>>(X4, P4C2, B, 256, 32 * 32, 384, 128);
  c2f_run(d_in, I_N4_C2F, P4C2, 384, 256, B, 32, 32, N4O, CAT, T1, T2, ar, stream);

  PackedConv n4c = pack_conv(d_in, I_N4_CONV, 256, 256, 3, ar, stream);
  conv_run(n4c, N4O, 256, 0, B, 32, 32, 2, P5C, 768, 0, ACT_SILU, stream);
  copy_into_kernel<<<cdiv(B * 512 * 16 * 16, 256), 256, 0, stream>>>(X5, P5C, B, 512, 16 * 16, 768, 256);
  c2f_run(d_in, I_N5_C2F, P5C, 768, 512, B, 16, 16, P5, CAT, T1, T2, ar, stream);

  // ---------------- detect heads ----------------
  float* out = (float*)d_out;
  const size_t off1 = 4ull * 64 * 64 * 9;
  const size_t off2 = off1 + 4ull * 32 * 32 * 9;
  det_run(d_in, I_DET0, P3,   128, 64, 64, out,        ar, B, stream);
  det_run(d_in, I_DET1, P4C2, 384, 32, 32, out + off1, ar, B, stream);
  det_run(d_in, I_DET2, P5,   512, 16, 16, out + off2, ar, B, stream);
}
